// BitLinear_29557964931127
// MI455X (gfx1250) — 8-bit matrix operand order repaired and checked by reading, compiled, not run on hardware
//
#include <hip/hip_runtime.h>
#include <stdint.h>

typedef __attribute__((ext_vector_type(8))) int v8i;
typedef __attribute__((ext_vector_type(4))) int v4i;
typedef __attribute__((ext_vector_type(4))) unsigned int v4u;

#define KD 4096
#define ND 4096
#define MD 32768
#define NK (KD / 64)

#define LROW 80
#define AW (256 * LROW)
#define BW (256 * LROW)

#if defined(__AMDGCN__) && defined(__HIP_DEVICE_COMPILE__)
#  if __has_builtin(__builtin_amdgcn_tensor_load_to_lds) && \
      __has_builtin(__builtin_amdgcn_s_wait_tensorcnt)
#    define USE_TDM 1
#  endif
#endif

__global__ __launch_bounds__(256) void w_reduce1(const float* __restrict__ w,
                                                 float* __restrict__ part) {
  __shared__ float s[256];
  const int tid = threadIdx.x;
  const size_t base = (size_t)blockIdx.x * 4096 + tid;
  float acc = 0.f;
#pragma unroll
  for (int i = 0; i < 16; ++i) acc += w[base + (size_t)i * 256];
  s[tid] = acc; __syncthreads();
  for (int off = 128; off > 0; off >>= 1) {
    if (tid < off) s[tid] += s[tid + off];
    __syncthreads();
  }
  if (tid == 0) part[blockIdx.x] = s[0];
}

__global__ __launch_bounds__(256) void w_reduce2(const float* __restrict__ part,
                                                 float* __restrict__ wmean) {
  __shared__ float s[256];
  const int tid = threadIdx.x;
  float acc = 0.f;
#pragma unroll
  for (int i = 0; i < 16; ++i) acc += part[tid + i * 256];
  s[tid] = acc; __syncthreads();
  for (int off = 128; off > 0; off >>= 1) {
    if (tid < off) s[tid] += s[tid + off];
    __syncthreads();
  }
  if (tid == 0) *wmean = s[0] * (1.0f / (4096.0f * 4096.0f));
}

__global__ __launch_bounds__(256) void rowquant_x(const float* __restrict__ x,
                                                  unsigned int* __restrict__ qA,
                                                  float* __restrict__ oscale) {
  __shared__ float s_ss[256];
  __shared__ float s_mx[256];
  __shared__ float s_qm;
  const int tid = threadIdx.x;
  const unsigned int m = blockIdx.x;
  const float4* xr = (const float4*)(x + (size_t)m * KD);
  float4 v[4];
  float ss = 0.f, mx = 0.f;
#pragma unroll
  for (int i = 0; i < 4; ++i) {
    v[i] = xr[tid + i * 256];
    ss += v[i].x * v[i].x + v[i].y * v[i].y + v[i].z * v[i].z + v[i].w * v[i].w;
    mx = fmaxf(mx, fmaxf(fmaxf(fabsf(v[i].x), fabsf(v[i].y)),
                         fmaxf(fabsf(v[i].z), fabsf(v[i].w))));
  }
  s_ss[tid] = ss; s_mx[tid] = mx; __syncthreads();
  for (int off = 128; off > 0; off >>= 1) {
    if (tid < off) {
      s_ss[tid] += s_ss[tid + off];
      s_mx[tid] = fmaxf(s_mx[tid], s_mx[tid + off]);
    }
    __syncthreads();
  }
  if (tid == 0) {
    float rsq  = rsqrtf(s_ss[0] * (1.0f / (float)KD) + 1e-5f);
    float maxn = fmaxf(s_mx[0] * rsq, 1e-5f);
    float xs   = 127.0f / maxn;
    s_qm = rsq * xs;
    oscale[m] = 1.0f / (xs * xs);
  }
  __syncthreads();
  const float qm = s_qm;
#pragma unroll
  for (int i = 0; i < 4; ++i) {
    int q0 = __float2int_rn(v[i].x * qm);
    int q1 = __float2int_rn(v[i].y * qm);
    int q2 = __float2int_rn(v[i].z * qm);
    int q3 = __float2int_rn(v[i].w * qm);
    q0 = max(-128, min(127, q0)); q1 = max(-128, min(127, q1));
    q2 = max(-128, min(127, q2)); q3 = max(-128, min(127, q3));
    unsigned int p = (unsigned int)(q0 & 0xff) | ((unsigned int)(q1 & 0xff) << 8) |
                     ((unsigned int)(q2 & 0xff) << 16) | ((unsigned int)(q3 & 0xff) << 24);
    const unsigned int k4 = (unsigned int)(tid + i * 256);
    const unsigned int dst = ((m >> 8) * 64u + (k4 >> 4)) * 4096u +
                             (m & 255u) * 16u + (k4 & 15u);
    qA[dst] = p;
  }
}

__global__ __launch_bounds__(256) void quant_w(const float* __restrict__ w,
                                               const float* __restrict__ wmean,
                                               unsigned int* __restrict__ qWt) {
  __shared__ float tile[64][65];
  const int tid = threadIdx.x;
  const int n0 = blockIdx.x * 64;
  const int k0 = blockIdx.y * 64;
  const float mean = *wmean;
#pragma unroll
  for (int i = 0; i < 16; ++i) {
    const int row = i * 4 + (tid >> 6);
    const int col = tid & 63;
    tile[row][col] = w[(size_t)(k0 + row) * ND + n0 + col];
  }
  __syncthreads();
  const int nl = tid >> 2;
  const int dg = tid & 3;
  const int n  = n0 + nl;
  const unsigned int base = ((unsigned int)(n >> 8) * 64u + (unsigned int)(k0 >> 6)) * 4096u +
                            (unsigned int)(n & 255) * 16u;
#pragma unroll
  for (int j = 0; j < 4; ++j) {
    const int dw = dg * 4 + j;
    unsigned int p = 0;
#pragma unroll
    for (int b = 0; b < 4; ++b) {
      float d = tile[dw * 4 + b][nl] - mean;
      int s = (d > 0.f) ? 1 : ((d < 0.f) ? -1 : 0);
      p |= (unsigned int)(s & 0xff) << (8 * b);
    }
    qWt[base + dw] = p;
  }
}

#if defined(USE_TDM)
__device__ __forceinline__ void tdm_load_panel(unsigned int lds_off, uint64_t gaddr) {
  v4u g0;
  g0.x = 1u;
  g0.y = lds_off;
  g0.z = (unsigned int)gaddr;
  g0.w = (unsigned int)(gaddr >> 32) | (2u << 30);
  v8i g1;
  g1[0] = (2 << 16) | (1 << 20) | (3 << 22) | (3 << 25);
  g1[1] = (16 << 16);
  g1[2] = (256 << 16);
  g1[3] = (16 << 16);
  g1[4] = 256;
  g1[5] = 16;
  g1[6] = 0;
  g1[7] = 0;
  v4i z4 = {0, 0, 0, 0};
#if __clang_major__ >= 23
  v8i z8 = {0, 0, 0, 0, 0, 0, 0, 0};
  __builtin_amdgcn_tensor_load_to_lds(g0, g1, z4, z4, z8, 0);
#else
  __builtin_amdgcn_tensor_load_to_lds(g0, g1, z4, z4, 0);
#endif
}
#endif

__global__ __launch_bounds__(512) void bitlinear_gemm(
    const signed char* __restrict__ qA, const signed char* __restrict__ qWt,
    const float* __restrict__ oscale, float* __restrict__ out) {
  __shared__ __align__(16) char lds[2][AW + BW];

  const int lane = threadIdx.x & 31;
  const int wave = threadIdx.x >> 5;
  const int wm = wave & 3;
  const int wn = wave >> 2;
  const int pm = blockIdx.y;
  const int pn = blockIdx.x;

  const signed char* apanels = qA  + (size_t)pm * 64 * 16384;
  const signed char* bpanels = qWt + (size_t)pn * 64 * 16384;

  v8i acc[4][4];
#pragma unroll
  for (int i = 0; i < 4; ++i)
#pragma unroll
    for (int j = 0; j < 4; ++j) acc[i][j] = (v8i){0, 0, 0, 0, 0, 0, 0, 0};

  auto compute_step = [&](const char* Ab, const char* Bb) {
    v8i af[4];
#pragma unroll
    for (int ms = 0; ms < 4; ++ms) {
      const int ml = wm * 64 + ms * 16 + (lane & 15);
      const int* pa = (const int*)(Ab + ml * LROW + ((lane >> 4) * 16));
      v8i a;
      a[0] = pa[0];  a[1] = pa[1];  a[2] = pa[2];  a[3] = pa[3];
      a[4] = pa[8];  a[5] = pa[9];  a[6] = pa[10]; a[7] = pa[11];
      af[ms] = a;
    }
#pragma unroll
    for (int ns = 0; ns < 4; ++ns) {
      const int nl = wn * 64 + ns * 16 + (lane & 15);
      const int4* pb = (const int4*)(Bb + nl * LROW + ((lane >> 4) * 16));
      int4 lo = pb[0];
      int4 hi = pb[2];
      v8i b;
      b[0] = lo.x; b[1] = lo.y; b[2] = lo.z; b[3] = lo.w;
      b[4] = hi.x; b[5] = hi.y; b[6] = hi.z; b[7] = hi.w;
#pragma unroll
      for (int ms = 0; ms < 4; ++ms)
        acc[ms][ns] = __builtin_amdgcn_wmma_i32_16x16x64_iu8(
            true, af[ms], true, b, acc[ms][ns], false, false);
    }
  };

#if defined(USE_TDM)
  if (wave == 0) {
    tdm_load_panel((unsigned int)(uintptr_t)&lds[0][0],  (uint64_t)(uintptr_t)apanels);
    tdm_load_panel((unsigned int)(uintptr_t)&lds[0][AW], (uint64_t)(uintptr_t)bpanels);
  }
  for (int ks = 0; ks < NK; ks += 2) {
    if (wave == 0) {
      tdm_load_panel((unsigned int)(uintptr_t)&lds[1][0],
                     (uint64_t)(uintptr_t)(apanels + (size_t)(ks + 1) * 16384));
      tdm_load_panel((unsigned int)(uintptr_t)&lds[1][AW],
                     (uint64_t)(uintptr_t)(bpanels + (size_t)(ks + 1) * 16384));
      __builtin_amdgcn_s_wait_tensorcnt(2);
    }
    __syncthreads();
    compute_step(&lds[0][0], &lds[0][AW]);
    __syncthreads();
    if (wave == 0) {
      if (ks + 2 < NK) {
        tdm_load_panel((unsigned int)(uintptr_t)&lds[0][0],
                       (uint64_t)(uintptr_t)(apanels + (size_t)(ks + 2) * 16384));
        tdm_load_panel((unsigned int)(uintptr_t)&lds[0][AW],
                       (uint64_t)(uintptr_t)(bpanels + (size_t)(ks + 2) * 16384));
        __builtin_amdgcn_s_wait_tensorcnt(2);
      } else {
        __builtin_amdgcn_s_wait_tensorcnt(0);
      }
    }
    __syncthreads();
    compute_step(&lds[1][0], &lds[1][AW]);
    __syncthreads();
  }
#else
  int4 ra[2]; int4 rb[2];
  auto load_regs = [&](int ks) {
    const int4* ga = (const int4*)(apanels + (size_t)ks * 16384);
    const int4* gb = (const int4*)(bpanels + (size_t)ks * 16384);
#pragma unroll
    for (int i = 0; i < 2; ++i) ra[i] = ga[i * 512 + threadIdx.x];
#pragma unroll
    for (int i = 0; i < 2; ++i) rb[i] = gb[i * 512 + threadIdx.x];
  };
  auto store_regs = [&](char* Ab, char* Bb) {
#pragma unroll
    for (int i = 0; i < 2; ++i) {
      int c = i * 512 + threadIdx.x;
      *(int4*)(Ab + (c >> 2) * LROW + (c & 3) * 16) = ra[i];
    }
#pragma unroll
    for (int i = 0; i < 2; ++i) {
      int c = i * 512 + threadIdx.x;
      *(int4*)(Bb + (c >> 2) * LROW + (c & 3) * 16) = rb[i];
    }
  };
  load_regs(0);
  store_regs(&lds[0][0], &lds[0][AW]);
  __syncthreads();
  for (int ks = 0; ks < NK; ks += 2) {
    load_regs(ks + 1);
    compute_step(&lds[0][0], &lds[0][AW]);
    __syncthreads();
    store_regs(&lds[1][0], &lds[1][AW]);
    __syncthreads();
    if (ks + 2 < NK) load_regs(ks + 2);
    compute_step(&lds[1][0], &lds[1][AW]);
    __syncthreads();
    if (ks + 2 < NK) {
      store_regs(&lds[0][0], &lds[0][AW]);
    }
    __syncthreads();
  }
#endif

  const int colBase = pn * 256 + wn * 64 + (lane & 15);
  const int rowBase = pm * 256 + wm * 64 + (lane >> 4) * 8;
#pragma unroll
  for (int ms = 0; ms < 4; ++ms) {
#pragma unroll
    for (int r = 0; r < 8; ++r) {
      const int row = rowBase + ms * 16 + r;
      const float sc = oscale[row];
      float* po = out + (size_t)row * ND + colBase;
#pragma unroll
      for (int ns = 0; ns < 4; ++ns)
        po[ns * 16] = (float)acc[ms][ns][r] * sc;
    }
  }
}

extern "C" void kernel_launch(void* const* d_in, const int* in_sizes, int n_in,
                              void* d_out, int out_size, void* d_ws, size_t ws_size,
                              hipStream_t stream) {
  const float* x = (const float*)d_in[0];
  const float* w = (const float*)d_in[1];
  float* out = (float*)d_out;

  char* ws = (char*)d_ws;
  signed char* qA   = (signed char*)ws;
  signed char* qWt  = (signed char*)(ws + (size_t)MD * KD);
  float* oscale = (float*)(ws + (size_t)MD * KD + (size_t)ND * KD);
  float* wpart  = oscale + MD;
  float* wmean  = wpart + 4096;

  w_reduce1<<<4096, 256, 0, stream>>>(w, wpart);
  w_reduce2<<<1, 256, 0, stream>>>(wpart, wmean);
  rowquant_x<<<MD, 256, 0, stream>>>(x, (unsigned int*)qA, oscale);
  {
    dim3 g(ND / 64, KD / 64);
    quant_w<<<g, 256, 0, stream>>>(w, wmean, (unsigned int*)qWt);
  }
  {
    dim3 g(ND / 256, MD / 256);
    bitlinear_gemm<<<g, 512, 0, stream>>>(qA, qWt, oscale, out);
  }
}
